// SelfAttention_40329742910051
// MI455X (gfx1250) — matrix operand order repaired and checked by reading, not compiled, not run on hardware
//
#include <hip/hip_runtime.h>
#include <hip/hip_bf16.h>
#include <math.h>

// ---------- types ----------
typedef __attribute__((ext_vector_type(16))) __bf16 v16bf;
typedef __attribute__((ext_vector_type(8)))  __bf16 v8bf;
typedef __attribute__((ext_vector_type(8)))  float  v8f;

#define BATCH  4
#define CCH    256
#define HEADS  4
#define HD     64
#define NSP    2304          // 48*48 tokens
#define GN_G   32            // C//8 groups
#define GSZ    8             // channels per group
#define GN_EPSF 1e-5f

__device__ __forceinline__ int lane_id() { return (int)(threadIdx.x & 31); }

// ---------------------------------------------------------------------------
// WMMA fragment loaders (bf16, 16x16x32 shape), per CDNA5 ISA layouts.
// A (16 rows x K=32): lane l -> row (l&15); elements 0..7 -> K = kb+0..7,
// elements 8..15 -> K = kb+16..23, kb = (l>=16 ? 8 : 0).
// ---------------------------------------------------------------------------
__device__ __forceinline__ v16bf load_a_frag(const __bf16* base, int ld,
                                             int row0, int k0) {
  int l = lane_id();
  const __bf16* p = base + (size_t)(row0 + (l & 15)) * ld + k0 + ((l >> 4) << 3);
  v8bf lo = *(const v8bf*)(p);        // K offsets +0..7
  v8bf hi = *(const v8bf*)(p + 16);   // K offsets +16..23
  v16bf r;
#pragma unroll
  for (int i = 0; i < 8; ++i) { r[i] = lo[i]; r[i + 8] = hi[i]; }
  return r;
}

// B (K=32 rows x 16 cols): lane l -> col (n0 + (l&15)); elements 0..7 -> K = kb+0..7,
// elements 8..15 -> K = kb+16..23, kb = (l>=16 ? 8 : 0).
__device__ __forceinline__ v16bf load_b_frag(const __bf16* base, int ld,
                                             int k0, int n0) {
  int l = lane_id();
  const __bf16* p = base + (size_t)(k0 + ((l >> 4) << 3)) * ld + n0 + (l & 15);
  v16bf r;
#pragma unroll
  for (int i = 0; i < 8; ++i) { r[i] = p[(size_t)i * ld]; r[i + 8] = p[(size_t)(16 + i) * ld]; }
  return r;
}

__device__ __forceinline__ v8f wmma_bf16(v16bf a, v16bf b, v8f c) {
  // (neg_a, A, neg_b, B, c_mod, C, reuse_a, reuse_b)
  return __builtin_amdgcn_wmma_f32_16x16x32_bf16(false, a, false, b,
                                                 (short)0, c, false, false);
}

// ---------------------------------------------------------------------------
// 0) f32 -> bf16 conversion
// ---------------------------------------------------------------------------
__global__ void k_cvt_bf16(const float* __restrict__ s, __bf16* __restrict__ d,
                           int n) {
  int i = blockIdx.x * blockDim.x + threadIdx.x;
  if (i < n) d[i] = (__bf16)s[i];
}

// ---------------------------------------------------------------------------
// 1) QKV GEMM: qkv[o,n] = sum_c W[o,c] * x[c,n] + b[o]
//    Epilogue scatters: Q -> q_t[b,h,n,d] (transposed), K,V -> [b,h,d,n] bf16
//    grid (NSP/128, 768/16, B), block 256 (8 waves, each a 16x16 tile)
// ---------------------------------------------------------------------------
__global__ __launch_bounds__(256) void k_qkv_gemm(
    const __bf16* __restrict__ xb, const __bf16* __restrict__ wb,
    const float* __restrict__ bias, __bf16* __restrict__ q_t,
    __bf16* __restrict__ kbuf, __bf16* __restrict__ vbuf) {
  int b  = blockIdx.z;
  int o0 = blockIdx.y * 16;
  int w  = threadIdx.x >> 5;
  int n0 = blockIdx.x * 128 + w * 16;
  int l  = lane_id();

  const __bf16* xB = xb + (size_t)b * CCH * NSP;
  v8f acc = {};
#pragma unroll
  for (int k0 = 0; k0 < CCH; k0 += 32) {
    v16bf a = load_a_frag(wb, CCH, o0, k0);
    v16bf bf = load_b_frag(xB, NSP, k0, n0);
    acc = wmma_bf16(a, bf, acc);
  }
  int n = n0 + (l & 15);
  int rofs = 8 * (l >> 4);
  if (o0 < CCH) {                       // ---- Q, store transposed [b,h,n,d]
    int h = o0 >> 6;
    __bf16* qrow = q_t + (((size_t)(b * HEADS + h)) * NSP + n) * HD;
#pragma unroll
    for (int r = 0; r < 8; ++r) {
      int o = o0 + r + rofs;
      qrow[o & 63] = (__bf16)(acc[r] + bias[o]);
    }
  } else if (o0 < 2 * CCH) {            // ---- K, store [b,h,d,n]
    int h = (o0 - CCH) >> 6;
    __bf16* kb = kbuf + ((size_t)(b * HEADS + h)) * HD * NSP;
#pragma unroll
    for (int r = 0; r < 8; ++r) {
      int o = o0 + r + rofs;
      kb[(size_t)((o - CCH) & 63) * NSP + n] = (__bf16)(acc[r] + bias[o]);
    }
  } else {                              // ---- V, store [b,h,d,n]
    int h = (o0 - 2 * CCH) >> 6;
    __bf16* vb = vbuf + ((size_t)(b * HEADS + h)) * HD * NSP;
#pragma unroll
    for (int r = 0; r < 8; ++r) {
      int o = o0 + r + rofs;
      vb[(size_t)((o - 2 * CCH) & 63) * NSP + n] = (__bf16)(acc[r] + bias[o]);
    }
  }
}

// ---------------------------------------------------------------------------
// 2) Attention, two-pass softmax, per-wave 16-row tile.
//    grid (NSP/64, B*HEADS), block 128 (4 waves)
// ---------------------------------------------------------------------------
__device__ __forceinline__ void attn_s_tile(const __bf16* Kp, v16bf aq0,
                                            v16bf aq1, int m, float* s) {
  v16bf blo = load_b_frag(Kp, NSP, 0,  m);   // d rows  0..31
  v16bf bhi = load_b_frag(Kp, NSP, 32, m);   // d rows 32..63
  v8f acc = {};
  acc = wmma_bf16(aq0, blo, acc);
  acc = wmma_bf16(aq1, bhi, acc);
#pragma unroll
  for (int r = 0; r < 8; ++r) s[r] = acc[r];
}

__global__ __launch_bounds__(128) void k_attn(
    const __bf16* __restrict__ q_t, const __bf16* __restrict__ kbuf,
    const __bf16* __restrict__ vbuf, __bf16* __restrict__ attnb) {
  __shared__ __align__(16) __bf16 lds_p[4][32][24];   // P^T staging per wave
  __shared__ float lds_sum[4][16];

  int bh = blockIdx.y;                   // b*HEADS + h
  int w  = threadIdx.x >> 5;
  int l  = lane_id();
  int n0 = blockIdx.x * 64 + w * 16;
  const float scale = 0.125f;            // hd^-0.5, hd=64

  const __bf16* Q = q_t + (size_t)bh * NSP * HD;
  const __bf16* K = kbuf + (size_t)bh * HD * NSP;
  const __bf16* V = vbuf + (size_t)bh * HD * NSP;

  v16bf aq0 = load_a_frag(Q, HD, n0, 0);
  v16bf aq1 = load_a_frag(Q, HD, n0, 32);

  // ---- pass 1: row max ----
  float mx[8];
#pragma unroll
  for (int r = 0; r < 8; ++r) mx[r] = -3.0e38f;
  for (int mc = 0; mc < NSP / 32; ++mc) {
    int m0 = mc * 32;
    float s0[8], s1[8];
    attn_s_tile(K, aq0, aq1, m0, s0);
    attn_s_tile(K, aq0, aq1, m0 + 16, s1);
#pragma unroll
    for (int r = 0; r < 8; ++r) mx[r] = fmaxf(mx[r], fmaxf(s0[r], s1[r]));
  }
#pragma unroll
  for (int off = 1; off <= 8; off <<= 1)
#pragma unroll
    for (int r = 0; r < 8; ++r)
      mx[r] = fmaxf(mx[r], __shfl_xor(mx[r], off, 32));

  // ---- pass 2: P = exp(scale*(S-mx)), O += V * P^T ----
  float rs[8];
#pragma unroll
  for (int r = 0; r < 8; ++r) rs[r] = 0.0f;
  v8f o[4];
#pragma unroll
  for (int dt = 0; dt < 4; ++dt) o[dt] = {};

  int ml = l & 15, nb = 8 * (l >> 4);
  for (int mc = 0; mc < NSP / 32; ++mc) {
    int m0 = mc * 32;
    if (mc + 1 < NSP / 32)               // hint next K rows toward WGP$
      __builtin_prefetch(K + (size_t)l * NSP + m0 + 32, 0, 1);
    float s0[8], s1[8];
    attn_s_tile(K, aq0, aq1, m0, s0);
    attn_s_tile(K, aq0, aq1, m0 + 16, s1);
#pragma unroll
    for (int r = 0; r < 8; ++r) {
      s0[r] = __expf(scale * (s0[r] - mx[r]));
      s1[r] = __expf(scale * (s1[r] - mx[r]));
      rs[r] += s0[r] + s1[r];
    }
    // stage P into LDS in B-fragment order: row m (=lane on write), cols n (=lane on reload)
#pragma unroll
    for (int r = 0; r < 8; ++r) {
      lds_p[w][ml][nb + r]      = (__bf16)s0[r];
      lds_p[w][ml + 16][nb + r] = (__bf16)s1[r];
    }
    // reload as B fragment: lane = n col (l&15), elements = m rows in table order
    int kb = 8 * (l >> 4);
    v16bf bp;
#pragma unroll
    for (int i = 0; i < 8; ++i) { bp[i] = lds_p[w][kb + i][ml]; bp[i + 8] = lds_p[w][16 + kb + i][ml]; }
#pragma unroll
    for (int dt = 0; dt < 4; ++dt) {
      v16bf av = load_a_frag(V, NSP, dt * 16, m0);   // rows = d, K = m
      o[dt] = wmma_bf16(av, bp, o[dt]);
    }
  }

  // ---- row sums -> LDS -> per-lane reciprocal ----
#pragma unroll
  for (int off = 1; off <= 8; off <<= 1)
#pragma unroll
    for (int r = 0; r < 8; ++r) rs[r] += __shfl_xor(rs[r], off, 32);
  if (l == 0)
#pragma unroll
    for (int r = 0; r < 8; ++r) lds_sum[w][r] = rs[r];
  if (l == 16)
#pragma unroll
    for (int r = 0; r < 8; ++r) lds_sum[w][8 + r] = rs[r];
  __syncthreads();
  float inv = 1.0f / lds_sum[w][l & 15];

  // ---- normalize + store bf16 [b, h*64+d, n] for proj GEMM ----
  int b = bh >> 2, h = bh & 3;
  int n = n0 + (l & 15), rofs = 8 * (l >> 4);
#pragma unroll
  for (int dt = 0; dt < 4; ++dt)
#pragma unroll
    for (int r = 0; r < 8; ++r) {
      int d = dt * 16 + r + rofs;
      attnb[((size_t)b * CCH + (h * HD + d)) * NSP + n] = (__bf16)(o[dt][r] * inv);
    }
}

// ---------------------------------------------------------------------------
// 3) Proj GEMM + bias + residual: y = proj_w @ attn + proj_b + x   (f32 out)
//    grid (NSP/128, 256/16, B), block 256
// ---------------------------------------------------------------------------
__global__ __launch_bounds__(256) void k_proj_gemm(
    const __bf16* __restrict__ attnb, const __bf16* __restrict__ wb,
    const float* __restrict__ bias, const float* __restrict__ xres,
    float* __restrict__ y) {
  int b  = blockIdx.z;
  int o0 = blockIdx.y * 16;
  int w  = threadIdx.x >> 5;
  int n0 = blockIdx.x * 128 + w * 16;
  int l  = lane_id();

  const __bf16* aB = attnb + (size_t)b * CCH * NSP;
  v8f acc = {};
#pragma unroll
  for (int k0 = 0; k0 < CCH; k0 += 32) {
    v16bf a = load_a_frag(wb, CCH, o0, k0);
    v16bf bf = load_b_frag(aB, NSP, k0, n0);
    acc = wmma_bf16(a, bf, acc);
  }
  int n = n0 + (l & 15), rofs = 8 * (l >> 4);
#pragma unroll
  for (int r = 0; r < 8; ++r) {
    int oc = o0 + r + rofs;
    size_t idx = ((size_t)b * CCH + oc) * NSP + n;
    y[idx] = acc[r] + bias[oc] + xres[idx];
  }
}

// ---------------------------------------------------------------------------
// 4) GroupNorm stats: per (b,g) mean & rstd over 8*2304 elems. grid 128x256
// ---------------------------------------------------------------------------
__global__ __launch_bounds__(256) void k_gn_stats(const float* __restrict__ y,
                                                  float* __restrict__ stats) {
  __shared__ float r1[256], r2[256];
  int bg = blockIdx.x, b = bg >> 5, g = bg & 31;
  float s = 0.0f, s2 = 0.0f;
#pragma unroll
  for (int ch = 0; ch < GSZ; ++ch) {
    const float* row = y + ((size_t)(b * CCH + g * GSZ + ch)) * NSP;
    for (int n = threadIdx.x; n < NSP; n += 256) {
      float v = row[n];
      s += v; s2 += v * v;
    }
  }
  r1[threadIdx.x] = s; r2[threadIdx.x] = s2;
  __syncthreads();
  for (int step = 128; step > 0; step >>= 1) {
    if ((int)threadIdx.x < step) {
      r1[threadIdx.x] += r1[threadIdx.x + step];
      r2[threadIdx.x] += r2[threadIdx.x + step];
    }
    __syncthreads();
  }
  if (threadIdx.x == 0) {
    const float cnt = (float)(GSZ * NSP);
    float mean = r1[0] / cnt;
    float var  = r2[0] / cnt - mean * mean;
    stats[2 * bg]     = mean;
    stats[2 * bg + 1] = rsqrtf(var + GN_EPSF);
  }
}

// 5) GroupNorm apply + affine -> d_out
__global__ __launch_bounds__(256) void k_gn_apply(
    const float* __restrict__ y, const float* __restrict__ stats,
    const float* __restrict__ gn_w, const float* __restrict__ gn_b,
    float* __restrict__ out, int total) {
  int i = blockIdx.x * blockDim.x + threadIdx.x;
  if (i >= total) return;
  int c = (i / NSP) & (CCH - 1);
  int b = i / (CCH * NSP);
  int bg = b * GN_G + (c >> 3);
  float mean = stats[2 * bg], rstd = stats[2 * bg + 1];
  out[i] = (y[i] - mean) * rstd * gn_w[c] + gn_b[c];
}

// ---------------------------------------------------------------------------
extern "C" void kernel_launch(void* const* d_in, const int* in_sizes, int n_in,
                              void* d_out, int out_size, void* d_ws,
                              size_t ws_size, hipStream_t stream) {
  const float* x      = (const float*)d_in[0];   // [4,256,48,48]
  const float* qkv_w  = (const float*)d_in[1];   // [768,256]
  const float* qkv_b  = (const float*)d_in[2];   // [768]
  const float* proj_w = (const float*)d_in[3];   // [256,256]
  const float* proj_b = (const float*)d_in[4];   // [256]
  const float* gn_w   = (const float*)d_in[5];   // [256]
  const float* gn_b   = (const float*)d_in[6];   // [256]
  float* out = (float*)d_out;

  // workspace carve-up (256B aligned)
  char* base = (char*)d_ws;
  size_t off = 0;
  auto alloc = [&](size_t bytes) {
    char* p = base + off;
    off = (off + bytes + 255) & ~(size_t)255;
    return p;
  };
  const size_t nBCN = (size_t)BATCH * CCH * NSP;        // 2,359,296
  __bf16* x_bf   = (__bf16*)alloc(nBCN * 2);
  __bf16* wq_bf  = (__bf16*)alloc((size_t)3 * CCH * CCH * 2);
  __bf16* wp_bf  = (__bf16*)alloc((size_t)CCH * CCH * 2);
  __bf16* q_t    = (__bf16*)alloc(nBCN * 2);
  __bf16* k_bf   = (__bf16*)alloc(nBCN * 2);
  __bf16* v_bf   = (__bf16*)alloc(nBCN * 2);
  __bf16* att_bf = (__bf16*)alloc(nBCN * 2);
  float*  y_ws   = (float*)alloc(nBCN * 4);
  float*  stats  = (float*)alloc((size_t)BATCH * GN_G * 2 * 4);
  (void)ws_size; (void)n_in; (void)in_sizes; (void)out_size;

  // 0) conversions
  k_cvt_bf16<<<(int)((nBCN + 255) / 256), 256, 0, stream>>>(x, x_bf, (int)nBCN);
  k_cvt_bf16<<<(3 * CCH * CCH + 255) / 256, 256, 0, stream>>>(qkv_w, wq_bf,
                                                              3 * CCH * CCH);
  k_cvt_bf16<<<(CCH * CCH + 255) / 256, 256, 0, stream>>>(proj_w, wp_bf,
                                                          CCH * CCH);
  // 1) QKV GEMM
  k_qkv_gemm<<<dim3(NSP / 128, 3 * CCH / 16, BATCH), 256, 0, stream>>>(
      x_bf, wq_bf, qkv_b, q_t, k_bf, v_bf);
  // 2) attention
  k_attn<<<dim3(NSP / 64, BATCH * HEADS), 128, 0, stream>>>(q_t, k_bf, v_bf,
                                                            att_bf);
  // 3) proj + residual
  k_proj_gemm<<<dim3(NSP / 128, CCH / 16, BATCH), 256, 0, stream>>>(
      att_bf, wp_bf, proj_b, x, y_ws);
  // 4/5) group norm
  k_gn_stats<<<BATCH * GN_G, 256, 0, stream>>>(y_ws, stats);
  k_gn_apply<<<(int)((nBCN + 255) / 256), 256, 0, stream>>>(
      y_ws, stats, gn_w, gn_b, out, (int)nBCN);
}
